// NonGatedGroupedExperts_28252294873411
// MI455X (gfx1250) — matrix operand order repaired and checked by reading, not compiled, not run on hardware
//
#include <hip/hip_runtime.h>

// ---------- problem constants (from reference) ----------
constexpr int E_ = 16;
constexpr int I_ = 1024;
constexpr int D_ = 2048;
constexpr int T_ = 32768;
constexpr int CAP = T_ / E_;     // 2048 per-expert capacity

// ---------- tiling ----------
constexpr int BM = 128;          // block tile M
constexpr int BN = 128;          // block tile N
constexpr int BK = 32;           // k-step (matches 16x16x32 WMMA)
constexpr int LDSP = BK + 8;     // padded LDS row stride (40 halfs = 80B, 16B aligned)

typedef _Float16 v16h __attribute__((ext_vector_type(16)));
typedef _Float16 v8h  __attribute__((ext_vector_type(8)));
typedef _Float16 v4h  __attribute__((ext_vector_type(4)));
typedef float    v8f  __attribute__((ext_vector_type(8)));
typedef float    v4f  __attribute__((ext_vector_type(4)));

__device__ __forceinline__ v16h cat8(v8h lo, v8h hi) {
    return __builtin_shufflevector(lo, hi, 0,1,2,3,4,5,6,7,8,9,10,11,12,13,14,15);
}

// A fragment (16x32, MxK) from LDS tile stored [m][k] with stride LDSP.
// Layout: lanes 0-15: M=lane, K=0..15;
//         lanes 16-31: M=lane-16, K=16..31.
__device__ __forceinline__ v16h load_frag_a(const _Float16* base, int lane) {
    const int r  = lane & 15;
    const int hi = lane >> 4;
    const _Float16* p = base + r * LDSP + hi * 16;
    v8h lo = *(const v8h*)(p);        // ds_load_b128
    v8h hh = *(const v8h*)(p + 8);    // ds_load_b128
    return cat8(lo, hh);
}

// B fragment (32x16, KxN) from LDS tile stored [n][k] with stride LDSP.
// Lanes 0-15 hold K=0..15 (2/VGPR), lanes 16-31 hold K=16..31; N = lane%16.
__device__ __forceinline__ v16h load_frag_b(const _Float16* base, int lane) {
    const int r  = lane & 15;
    const int hi = lane >> 4;
    const _Float16* p = base + r * LDSP + hi * 16;
    v8h lo = *(const v8h*)(p);        // ds_load_b128
    v8h hh = *(const v8h*)(p + 8);    // ds_load_b128
    return cat8(lo, hh);
}

__device__ __forceinline__ int expert_start(const int* __restrict__ counts, int e) {
    int s = 0;
    for (int i = 0; i < e; ++i) s += counts[i];   // uniform -> scalar loads
    return s;
}

// ================= GEMM1: H[e] = relu(X_e @ W1_e^T)^2, f16 out =================
// Note: H rows >= count are never used by stored outputs (GEMM2 guards its
// stores), so A rows are merely CLAMPED for address safety -- no masking.
__global__ __launch_bounds__(256)
void moe_gemm1(const float* __restrict__ x, const float* __restrict__ w1,
               _Float16* __restrict__ Hws, const int* __restrict__ counts)
{
    const int e  = blockIdx.z;
    const int bm = blockIdx.y;
    const int bn = blockIdx.x;

    const int start = expert_start(counts, e);
    int nval = counts[e]; if (nval > CAP) nval = CAP;
    const int rmax = nval > 0 ? nval - 1 : 0;

    __shared__ __align__(16) _Float16 Alds[BM * LDSP];
    __shared__ __align__(16) _Float16 Blds[BN * LDSP];

    const int tid  = threadIdx.x;
    const int lane = tid & 31;
    const int wave = tid >> 5;         // 0..7
    const int mo   = (wave & 3) * 32;  // 4 waves along M
    const int no   = (wave >> 2) * 64; // 2 waves along N

    v8f acc[2][4] = {};

    const float* Abase = x  + (size_t)start * D_;
    const float* Bbase = w1 + (size_t)e * I_ * D_;

    // --- hoisted staging addresses (k-invariant) ---
    const float* aptr[4]; const float* bptr[4];
    unsigned sA[4], sB[4];
    #pragma unroll
    for (int j = 0; j < 4; ++j) {
        const int lin = j * 256 + tid;       // float4 chunk id (1024 total)
        const int row = lin >> 3;
        const int col = (lin & 7) << 2;
        int gr = bm * BM + row;
        if (gr > rmax) gr = rmax;            // clamp, no masking needed
        aptr[j] = Abase + (size_t)gr * D_ + col;
        bptr[j] = Bbase + (size_t)(bn * BN + row) * D_ + col;
        sA[j] = row * LDSP + col;
        sB[j] = row * LDSP + col;
    }

    for (int kk = 0; kk < D_; kk += BK) {
        __syncthreads();
        #pragma unroll
        for (int j = 0; j < 4; ++j) {
            v4f v = *(const v4f*)(aptr[j] + kk);
            v4h h = { (_Float16)v.x, (_Float16)v.y, (_Float16)v.z, (_Float16)v.w };
            *(v4h*)(&Alds[sA[j]]) = h;
        }
        #pragma unroll
        for (int j = 0; j < 4; ++j) {
            v4f v = *(const v4f*)(bptr[j] + kk);
            v4h h = { (_Float16)v.x, (_Float16)v.y, (_Float16)v.z, (_Float16)v.w };
            *(v4h*)(&Blds[sB[j]]) = h;
        }
        __syncthreads();

        v16h af[2], bf[4];
        #pragma unroll
        for (int i = 0; i < 2; ++i) af[i] = load_frag_a(&Alds[(mo + 16 * i) * LDSP], lane);
        #pragma unroll
        for (int j = 0; j < 4; ++j) bf[j] = load_frag_b(&Blds[(no + 16 * j) * LDSP], lane);

        #pragma unroll
        for (int i = 0; i < 2; ++i)
            #pragma unroll
            for (int j = 0; j < 4; ++j)
                acc[i][j] = __builtin_amdgcn_wmma_f32_16x16x32_f16(
                    false, af[i], false, bf[j], (short)0, acc[i][j], false, false);
    }

    // --- epilogue: relu^2, store f16 H ---
    _Float16* Hexp = Hws + (size_t)e * CAP * I_;
    const int hi = lane >> 4, nl = lane & 15;
    #pragma unroll
    for (int i = 0; i < 2; ++i)
        #pragma unroll
        for (int j = 0; j < 4; ++j)
            #pragma unroll
            for (int v = 0; v < 8; ++v) {
                const int m = bm * BM + mo + 16 * i + v + hi * 8;
                const int n = bn * BN + no + 16 * j + nl;
                float f = acc[i][j][v];
                f = f > 0.f ? f * f : 0.f;
                Hexp[(size_t)m * I_ + n] = (_Float16)f;
            }
}

// ================= GEMM2: Y = H[e] @ W2_e^T, f32 out to token rows =============
__global__ __launch_bounds__(256)
void moe_gemm2(const _Float16* __restrict__ Hws, const float* __restrict__ w2,
               float* __restrict__ out, const int* __restrict__ counts)
{
    const int e  = blockIdx.z;
    const int bm = blockIdx.y;
    const int bn = blockIdx.x;

    const int start = expert_start(counts, e);
    int nval = counts[e]; if (nval > CAP) nval = CAP;

    __shared__ __align__(16) _Float16 Alds[BM * LDSP];
    __shared__ __align__(16) _Float16 Blds[BN * LDSP];

    const int tid  = threadIdx.x;
    const int lane = tid & 31;
    const int wave = tid >> 5;
    const int mo   = (wave & 3) * 32;
    const int no   = (wave >> 2) * 64;

    v8f acc[2][4] = {};

    const _Float16* Hexp  = Hws + (size_t)e * CAP * I_;
    const float*    Bbase = w2 + (size_t)e * D_ * I_;

    // --- hoisted staging addresses ---
    // A tile (f16 H): async copy global -> LDS, 16B chunks, 2 per thread.
    const _Float16* gA[2]; unsigned ldsA[2];
    #pragma unroll
    for (int j = 0; j < 2; ++j) {
        const int lin = j * 256 + tid;       // 16B chunk id (512 total)
        const int row = lin >> 2;
        const int col = (lin & 3) << 3;
        gA[j]  = Hexp + (size_t)(bm * BM + row) * I_ + col;
        // low 32 bits of the flat shared address == LDS byte offset (ISA 10.2)
        ldsA[j] = (unsigned)(uintptr_t)(&Alds[row * LDSP + col]);
    }
    const float* bptr[4]; unsigned sB[4];
    #pragma unroll
    for (int j = 0; j < 4; ++j) {
        const int lin = j * 256 + tid;
        const int row = lin >> 3;
        const int col = (lin & 7) << 2;
        bptr[j] = Bbase + (size_t)(bn * BN + row) * I_ + col;
        sB[j] = row * LDSP + col;
    }

    for (int kk = 0; kk < I_; kk += BK) {
        __syncthreads();
        // --- A tile: async global->LDS (no VGPR round trip, ASYNCcnt) ---
        #pragma unroll
        for (int j = 0; j < 2; ++j) {
            asm volatile("global_load_async_to_lds_b128 %0, %1, off"
                         :: "v"(ldsA[j]), "v"(gA[j] + kk) : "memory");
        }
        // --- B tile: w2 f32 -> f16 staging ---
        #pragma unroll
        for (int j = 0; j < 4; ++j) {
            v4f v = *(const v4f*)(bptr[j] + kk);
            v4h h = { (_Float16)v.x, (_Float16)v.y, (_Float16)v.z, (_Float16)v.w };
            *(v4h*)(&Blds[sB[j]]) = h;
        }
        asm volatile("s_wait_asynccnt 0" ::: "memory");
        __syncthreads();

        v16h af[2], bf[4];
        #pragma unroll
        for (int i = 0; i < 2; ++i) af[i] = load_frag_a(&Alds[(mo + 16 * i) * LDSP], lane);
        #pragma unroll
        for (int j = 0; j < 4; ++j) bf[j] = load_frag_b(&Blds[(no + 16 * j) * LDSP], lane);

        #pragma unroll
        for (int i = 0; i < 2; ++i)
            #pragma unroll
            for (int j = 0; j < 4; ++j)
                acc[i][j] = __builtin_amdgcn_wmma_f32_16x16x32_f16(
                    false, af[i], false, bf[j], (short)0, acc[i][j], false, false);
    }

    // --- epilogue: write f32 output rows for real tokens only ---
    const int hi = lane >> 4, nl = lane & 15;
    #pragma unroll
    for (int i = 0; i < 2; ++i)
        #pragma unroll
        for (int j = 0; j < 4; ++j)
            #pragma unroll
            for (int v = 0; v < 8; ++v) {
                const int m = bm * BM + mo + 16 * i + v + hi * 8;
                const int n = bn * BN + no + 16 * j + nl;
                if (m < nval)
                    out[(size_t)(start + m) * D_ + n] = acc[i][j][v];
            }
}

extern "C" void kernel_launch(void* const* d_in, const int* in_sizes, int n_in,
                              void* d_out, int out_size, void* d_ws, size_t ws_size,
                              hipStream_t stream) {
    (void)in_sizes; (void)n_in; (void)out_size; (void)ws_size;
    const float* x      = (const float*)d_in[0];
    const float* w1     = (const float*)d_in[1];
    const float* w2     = (const float*)d_in[2];
    const int*   counts = (const int*)d_in[3];
    float* out = (float*)d_out;
    _Float16* Hws = (_Float16*)d_ws;   // needs E*CAP*I*2 = 64 MiB

    // zero output so dropped / padding tokens match the reference
    hipMemsetAsync(d_out, 0, (size_t)T_ * D_ * sizeof(float), stream);

    dim3 blk(256);
    dim3 g1(I_ / BN, CAP / BM, E_);   // 8 x 16 x 16
    moe_gemm1<<<g1, blk, 0, stream>>>(x, w1, Hws, counts);
    dim3 g2(D_ / BN, CAP / BM, E_);   // 16 x 16 x 16
    moe_gemm2<<<g2, blk, 0, stream>>>(Hws, w2, out, counts);
}
